// ViT_MultiTask_63118839382437
// MI455X (gfx1250) — matrix operand order repaired and checked by reading, not compiled, not run on hardware
//
#include <hip/hip_runtime.h>
#include <hip/hip_bf16.h>
#include <math.h>

typedef __attribute__((ext_vector_type(16))) _Float16 v16h;
typedef __attribute__((ext_vector_type(8)))  _Float16 v8h;
typedef __attribute__((ext_vector_type(8)))  float    v8f;

#define LAYERS 6
#define NHEADS 4
#define DMODEL 128
#define HD     32
#define FFDIM  512
#define C_MID  8
#define C_OUT  16
#define Q_OUT  10
#define MAXLEN 512
#define NCLS   3
#define BATCH  32
#define SEQ    500
#define FIN    232
#define NTOK   (BATCH*SEQ)   /* 16000 */
#define TPAD   512

// ---------------------------------------------------------------------------
// One-shot f32 -> f16 weight conversion (weights are L2-resident afterwards;
// amortized over 24 GEMM launches that each re-read them 125x).
// ---------------------------------------------------------------------------
__global__ __launch_bounds__(256)
void cvt_f32_to_f16_kernel(const float* __restrict__ in,
                           _Float16* __restrict__ out, int n)
{
    int i = blockIdx.x * 256 + threadIdx.x;
    if (i < n) out[i] = (_Float16)in[i];
}

// ---------------------------------------------------------------------------
// Encoder + token embedding: per (b,t) position, all-scalar (tensors too small
// for 16x16 WMMA tiles).  One 256-thread block per position.
// Writes both f32 h (residual path) and f16 h16 (WMMA A operand).
// ---------------------------------------------------------------------------
__global__ __launch_bounds__(256)
void encoder_embed_kernel(const float* __restrict__ x1,
                          const float* __restrict__ pw,  const float* __restrict__ pb,
                          const float* __restrict__ qw,
                          const float* __restrict__ fw,  const float* __restrict__ fb,
                          const float* __restrict__ ew,  const float* __restrict__ eb,
                          float* __restrict__ hout, _Float16* __restrict__ hout16)
{
    __shared__ float xs[FIN];
    __shared__ float xp[FIN * C_MID];
    __shared__ float sc[Q_OUT * FIN];
    __shared__ float red[256];
    __shared__ float qsum[Q_OUT];
    __shared__ float agg[Q_OUT * C_MID];
    __shared__ float tok[C_OUT * Q_OUT];

    const int b = blockIdx.x / SEQ;
    const int t = blockIdx.x % SEQ;
    const int tid = threadIdx.x;

    const float* xr = x1 + ((size_t)b * SEQ + t) * FIN;
    for (int f = tid; f < FIN; f += 256) xs[f] = xr[f];
    __syncthreads();

    for (int id = tid; id < FIN * C_MID; id += 256) {
        int f = id / C_MID, c = id % C_MID;
        float pe = (c & 1) ? 1.0f : 0.0f;
        xp[id] = xs[f] * pw[c] + pb[c] + pe;
    }
    __syncthreads();

    for (int id = tid; id < Q_OUT * FIN; id += 256) {
        int q = id / FIN, f = id % FIN;
        float s = 0.f;
        #pragma unroll
        for (int c = 0; c < C_MID; ++c) s += xp[f * C_MID + c] * qw[q * C_MID + c];
        sc[id] = s;
    }
    __syncthreads();

    for (int q = 0; q < Q_OUT; ++q) {
        float v = (tid < FIN) ? sc[q * FIN + tid] : -1e30f;
        red[tid] = v; __syncthreads();
        for (int s = 128; s > 0; s >>= 1) {
            if (tid < s) red[tid] = fmaxf(red[tid], red[tid + s]);
            __syncthreads();
        }
        float mx = red[0]; __syncthreads();
        float e = 0.f;
        if (tid < FIN) { e = __expf(sc[q * FIN + tid] - mx); sc[q * FIN + tid] = e; }
        red[tid] = e; __syncthreads();
        for (int s = 128; s > 0; s >>= 1) {
            if (tid < s) red[tid] += red[tid + s];
            __syncthreads();
        }
        if (tid == 0) qsum[q] = red[0];
        __syncthreads();
    }

    for (int id = tid; id < Q_OUT * C_MID; id += 256) {
        int q = id / C_MID, c = id % C_MID;
        float s = 0.f;
        for (int f = 0; f < FIN; ++f) s += sc[q * FIN + f] * xp[f * C_MID + c];
        agg[id] = s / qsum[q];
    }
    __syncthreads();

    for (int id = tid; id < C_OUT * Q_OUT; id += 256) {
        int o = id / Q_OUT, q = id % Q_OUT;
        float s = fb[o];
        #pragma unroll
        for (int c = 0; c < C_MID; ++c) s += agg[q * C_MID + c] * fw[c * C_OUT + o];
        tok[id] = s;
    }
    __syncthreads();

    if (tid < DMODEL) {
        float s = eb[tid];
        for (int j = 0; j < C_OUT * Q_OUT; ++j)
            s += tok[j] * ew[(size_t)j * DMODEL + tid];
        size_t o = ((size_t)b * SEQ + t) * DMODEL + tid;
        hout[o]   = s;
        hout16[o] = (_Float16)s;
    }
}

// ---------------------------------------------------------------------------
// Panel GEMM: C16[M,N] = f16( A16[M,K] @ B16[K,N] + bias ) (+relu)
//   A: f16 activations.  B: f16 pre-converted weights.  Both row-major.
// Block = 256 threads = 8 waves = 128 rows x 16 cols output panel.
// Per 32-K chunk the 32x16 weight tile is staged once into LDS transposed
// ([col][k]) and shared by all 8 waves; each wave feeds v_wmma from two
// ds b128 loads + two global b128 A loads.
// A lane layout: lanes 0-15 row m, K{0..7,16..23}; lanes 16-31 K{8..15,24..31}.
// B lane layout: lane%16 = column, K{0..7,16..23}/{8..15,24..31} split by lane half.
// ---------------------------------------------------------------------------
__global__ __launch_bounds__(256)
void gemm_f16_wmma(const _Float16* __restrict__ A, const _Float16* __restrict__ B,
                   const float* __restrict__ bias, _Float16* __restrict__ C,
                   int M, int N, int K, int relu)
{
    __shared__ _Float16 Blds[16 * 32];   // [col][k] for current K chunk

    const int tilesN = N >> 4;
    const int bm0 = (blockIdx.x / tilesN) * 128;
    const int n0  = (blockIdx.x % tilesN) * 16;
    const int tid  = threadIdx.x;
    const int wave = tid >> 5;
    const int lane = tid & 31;
    const int l16  = lane & 15;
    const int hi   = lane >> 4;

    const _Float16* Arow = A + (size_t)(bm0 + wave * 16 + l16) * K;

    // staging assignment: each thread copies 2 adjacent halfs of the chunk
    const int sid = tid * 2;            // 0..510
    const int sk  = sid >> 4;           // k row within chunk (0..31)
    const int sn  = sid & 15;           // col pair base (even)

    v8f acc = {};
    for (int k0 = 0; k0 < K; k0 += 32) {
        // prefetch next A chunk (global_prefetch_b8; speculative, OOB dropped)
        __builtin_prefetch((const void*)(Arow + k0 + 32), 0, 1);

        // ---- cooperative weight staging: 512 f16, coalesced, transposed ----
        {
            const _Float16* src = B + (size_t)(k0 + sk) * N + n0 + sn;
            Blds[sn * 32 + sk]       = src[0];
            Blds[(sn + 1) * 32 + sk] = src[1];
        }
        __syncthreads();

        v16h a, bm;
        {
            v8h alo = *(const v8h*)(Arow + k0 + hi * 8);
            v8h ahi = *(const v8h*)(Arow + k0 + hi * 8 + 16);
            #pragma unroll
            for (int i = 0; i < 8; ++i) { a[i] = alo[i]; a[i + 8] = ahi[i]; }
        }
        {
            const v8h* bp = (const v8h*)(Blds + l16 * 32 + hi * 8);
            v8h b0 = bp[0], b1 = bp[2];
            #pragma unroll
            for (int i = 0; i < 8; ++i) { bm[i] = b0[i]; bm[i + 8] = b1[i]; }
        }
        acc = __builtin_amdgcn_wmma_f32_16x16x32_f16(false, a, false, bm,
                                                     (short)0, acc, false, false);
        __syncthreads();
    }

    const int n = n0 + l16;
    const float bv = bias[n];
    #pragma unroll
    for (int r = 0; r < 8; ++r) {
        int m = bm0 + wave * 16 + r + hi * 8;
        float v = acc[r] + bv;
        if (relu) v = v > 0.f ? v : 0.f;
        C[(size_t)m * N + n] = (_Float16)v;
    }
}

// ---------------------------------------------------------------------------
// Fused attention: one wave per (b, head, 16-row query block).
// qkv is f16, rows of 384: [q|k|v].  S strip (16x512 f32) lives in LDS;
// softmax renormalization is folded into the A-tile build for P@V (the
// A row is always lane&15, so 1/rowsum is a per-lane scalar).
// Output written as f16 (feeds the WO GEMM directly).
// ---------------------------------------------------------------------------
__global__ __launch_bounds__(32)
void attn_wmma(const _Float16* __restrict__ qkv, const float* __restrict__ rel,
               _Float16* __restrict__ out)
{
    __shared__ float S[16 * TPAD];   // 32 KB
    __shared__ float red[32];

    const int qb = blockIdx.x & 31;
    const int h  = (blockIdx.x >> 5) & 3;
    const int b  = blockIdx.x >> 7;
    const int lane = threadIdx.x;
    const int l16  = lane & 15;
    const int hi   = lane >> 4;
    const size_t baseTok = (size_t)b * SEQ;
    const float scale = 0.17677669529663687f;   // 1/sqrt(32)

    // --- A tile: 16 query rows x hd(32), two aligned b128 loads per lane ---
    const int qrow  = qb * 16 + l16;
    const int qclmp = qrow < SEQ ? qrow : SEQ - 1;
    const _Float16* qp = qkv + (baseTok + qclmp) * 384 + h * HD;
    v16h a;
    {
        v8h qlo = *(const v8h*)(qp + hi * 8);
        v8h qhi = *(const v8h*)(qp + hi * 8 + 16);
        #pragma unroll
        for (int i = 0; i < 8; ++i) { a[i] = qlo[i]; a[i + 8] = qhi[i]; }
    }

    // --- score strip: 32 column tiles of 16 ---
    for (int ct = 0; ct < 32; ++ct) {
        const int col  = ct * 16 + l16;
        const int colc = col < SEQ ? col : SEQ - 1;
        const _Float16* kp = qkv + (baseTok + colc) * 384 + 128 + h * HD;
        v16h bm;
        {
            v8h k0 = *(const v8h*)(kp + hi * 8);
            v8h k1 = *(const v8h*)(kp + hi * 8 + 16);
            #pragma unroll
            for (int i = 0; i < 8; ++i) { bm[i] = k0[i]; bm[i + 8] = k1[i]; }
        }
        v8f acc = {};
        acc = __builtin_amdgcn_wmma_f32_16x16x32_f16(false, a, false, bm,
                                                     (short)0, acc, false, false);
        #pragma unroll
        for (int r = 0; r < 8; ++r) {
            int m    = r + hi * 8;
            int cc   = ct * 16 + l16;
            int grow = qb * 16 + m;
            float v;
            if (cc < SEQ) {
                int idx = cc - grow + (MAXLEN - 1);
                idx = idx < 0 ? 0 : (idx > 2 * MAXLEN - 2 ? 2 * MAXLEN - 2 : idx);
                v = acc[r] * scale + rel[idx];
            } else {
                v = -1e30f;
            }
            S[m * TPAD + cc] = v;
        }
    }
    __syncthreads();

    // --- softmax: lane pair (l16, l16+16) shares row l16, 256 cols each ---
    float inv;
    {
        const int row = l16;
        const int c0  = hi * 256;
        float m1 = -1e30f;
        for (int c = c0; c < c0 + 256; ++c) m1 = fmaxf(m1, S[row * TPAD + c]);
        red[lane] = m1; __syncthreads();
        float rowmax = fmaxf(red[row], red[row + 16]);
        __syncthreads();
        float s1 = 0.f;
        for (int c = c0; c < c0 + 256; ++c) {
            float e = __expf(S[row * TPAD + c] - rowmax);
            S[row * TPAD + c] = e;
            s1 += e;
        }
        red[lane] = s1; __syncthreads();
        inv = 1.f / (red[row] + red[row + 16]);   // same for both halves of pair
    }
    __syncthreads();

    // --- O = P(16x512) @ V(512x32); P built on the fly from S * inv ---
    v8f acc0 = {}, acc1 = {};
    for (int k0 = 0; k0 < TPAD; k0 += 32) {
        v16h pa;
        #pragma unroll
        for (int i = 0; i < 8; ++i) {
            pa[i]     = (_Float16)(S[l16 * TPAD + k0 + hi * 16 + i]      * inv);
            pa[i + 8] = (_Float16)(S[l16 * TPAD + k0 + hi * 16 + 8 + i] * inv);
        }
        v16h vb0, vb1;
        #pragma unroll
        for (int i = 0; i < 16; ++i) {
            int kk = k0 + hi * 16 + i;
            int kc = kk < SEQ ? kk : SEQ - 1;          // P==0 there anyway
            const _Float16* vp = qkv + (baseTok + kc) * 384 + 256 + h * HD;
            vb0[i] = vp[l16];
            vb1[i] = vp[16 + l16];
        }
        acc0 = __builtin_amdgcn_wmma_f32_16x16x32_f16(false, pa, false, vb0,
                                                      (short)0, acc0, false, false);
        acc1 = __builtin_amdgcn_wmma_f32_16x16x32_f16(false, pa, false, vb1,
                                                      (short)0, acc1, false, false);
    }
    #pragma unroll
    for (int r = 0; r < 8; ++r) {
        int m    = r + hi * 8;
        int grow = qb * 16 + m;
        if (grow < SEQ) {
            _Float16* op = out + (baseTok + grow) * DMODEL + h * HD;
            op[l16]      = (_Float16)acc0[r];
            op[16 + l16] = (_Float16)acc1[r];
        }
    }
}

// ---------------------------------------------------------------------------
// h = LayerNorm(h + r16) * g + b ; also refreshes the f16 mirror h16.
// ---------------------------------------------------------------------------
__global__ __launch_bounds__(128)
void add_ln_kernel(float* __restrict__ h, _Float16* __restrict__ h16,
                   const _Float16* __restrict__ r,
                   const float* __restrict__ g, const float* __restrict__ bb)
{
    __shared__ float red[128];
    const int row = blockIdx.x, d = threadIdx.x;
    const size_t o = (size_t)row * DMODEL + d;
    float x = h[o] + (float)r[o];
    red[d] = x; __syncthreads();
    for (int s = 64; s > 0; s >>= 1) { if (d < s) red[d] += red[d + s]; __syncthreads(); }
    float mean = red[0] * (1.f / DMODEL); __syncthreads();
    float c = x - mean;
    red[d] = c * c; __syncthreads();
    for (int s = 64; s > 0; s >>= 1) { if (d < s) red[d] += red[d + s]; __syncthreads(); }
    float var = red[0] * (1.f / DMODEL);
    float y = c * rsqrtf(var + 1e-5f) * g[d] + bb[d];
    h[o]   = y;
    h16[o] = (_Float16)y;
}

// ---------------------------------------------------------------------------
// mean-pool over T, final LN, 3-class head.  One block per batch element.
// ---------------------------------------------------------------------------
__global__ __launch_bounds__(128)
void pool_cls_kernel(const float* __restrict__ h,
                     const float* __restrict__ g, const float* __restrict__ bb,
                     const float* __restrict__ cw, const float* __restrict__ cb,
                     float* __restrict__ out)
{
    __shared__ float red[128];
    __shared__ float py[128];
    const int b = blockIdx.x, d = threadIdx.x;
    float s = 0.f;
    for (int t = 0; t < SEQ; ++t) s += h[((size_t)b * SEQ + t) * DMODEL + d];
    float x = s * (1.f / SEQ);
    red[d] = x; __syncthreads();
    for (int st = 64; st > 0; st >>= 1) { if (d < st) red[d] += red[d + st]; __syncthreads(); }
    float mean = red[0] * (1.f / DMODEL); __syncthreads();
    float c = x - mean;
    red[d] = c * c; __syncthreads();
    for (int st = 64; st > 0; st >>= 1) { if (d < st) red[d] += red[d + st]; __syncthreads(); }
    float var = red[0] * (1.f / DMODEL);
    py[d] = c * rsqrtf(var + 1e-5f) * g[d] + bb[d];
    __syncthreads();
    if (d < NCLS) {
        float acc = cb[d];
        for (int k = 0; k < DMODEL; ++k) acc += py[k] * cw[k * NCLS + d];
        out[b * NCLS + d] = acc;
    }
}

// ---------------------------------------------------------------------------
extern "C" void kernel_launch(void* const* d_in, const int* in_sizes, int n_in,
                              void* d_out, int out_size, void* d_ws, size_t ws_size,
                              hipStream_t stream)
{
    (void)in_sizes; (void)n_in; (void)out_size; (void)ws_size;

    const float* x1      = (const float*)d_in[0];
    const float* proj_w  = (const float*)d_in[1];
    const float* proj_b  = (const float*)d_in[2];
    const float* queries = (const float*)d_in[3];
    const float* fc_w    = (const float*)d_in[4];
    const float* fc_b    = (const float*)d_in[5];
    const float* embed_w = (const float*)d_in[6];
    const float* embed_b = (const float*)d_in[7];
    const float* wqkv    = (const float*)d_in[8];
    const float* bqkv    = (const float*)d_in[9];
    const float* wo      = (const float*)d_in[10];
    const float* bo      = (const float*)d_in[11];
    const float* w1      = (const float*)d_in[12];
    const float* b1      = (const float*)d_in[13];
    const float* w2      = (const float*)d_in[14];
    const float* b2      = (const float*)d_in[15];
    const float* ln1_g   = (const float*)d_in[16];
    const float* ln1_b   = (const float*)d_in[17];
    const float* ln2_g   = (const float*)d_in[18];
    const float* ln2_b   = (const float*)d_in[19];
    const float* rel     = (const float*)d_in[20];
    const float* final_g = (const float*)d_in[21];
    const float* final_b = (const float*)d_in[22];
    const float* cls_w   = (const float*)d_in[23];
    const float* cls_b   = (const float*)d_in[24];

    // workspace layout:
    //   f32 h | f16 h16 | f16 bufA (qkv / ff-mid) | f16 bufB | f16 weights x4
    float*     h      = (float*)d_ws;
    _Float16*  h16    = (_Float16*)(h + (size_t)NTOK * DMODEL);
    _Float16*  bufA   = h16  + (size_t)NTOK * DMODEL;
    _Float16*  bufB   = bufA + (size_t)NTOK * FFDIM;
    _Float16*  wqkv16 = bufB + (size_t)NTOK * DMODEL;
    _Float16*  wo16   = wqkv16 + (size_t)LAYERS * DMODEL * 3 * DMODEL;
    _Float16*  w116   = wo16   + (size_t)LAYERS * DMODEL * DMODEL;
    _Float16*  w216   = w116   + (size_t)LAYERS * DMODEL * FFDIM;

    // ---- one-shot weight conversion (2.4 MB f16 total, L2-resident) ----
    {
        int n;
        n = LAYERS * DMODEL * 3 * DMODEL;
        cvt_f32_to_f16_kernel<<<(n + 255) / 256, 256, 0, stream>>>(wqkv, wqkv16, n);
        n = LAYERS * DMODEL * DMODEL;
        cvt_f32_to_f16_kernel<<<(n + 255) / 256, 256, 0, stream>>>(wo, wo16, n);
        n = LAYERS * DMODEL * FFDIM;
        cvt_f32_to_f16_kernel<<<(n + 255) / 256, 256, 0, stream>>>(w1, w116, n);
        n = LAYERS * FFDIM * DMODEL;
        cvt_f32_to_f16_kernel<<<(n + 255) / 256, 256, 0, stream>>>(w2, w216, n);
    }

    encoder_embed_kernel<<<NTOK, 256, 0, stream>>>(
        x1, proj_w, proj_b, queries, fc_w, fc_b, embed_w, embed_b, h, h16);

    const int mp = NTOK / 128;   // 125 row panels
    for (int l = 0; l < LAYERS; ++l) {
        // qkv = h @ wqkv[l] + bqkv[l]        (16000x128 @ 128x384) -> f16
        gemm_f16_wmma<<<mp * (3 * DMODEL / 16), 256, 0, stream>>>(
            h16, wqkv16 + (size_t)l * DMODEL * 3 * DMODEL, bqkv + l * 3 * DMODEL,
            bufA, NTOK, 3 * DMODEL, DMODEL, 0);

        // fused attention -> bufB (16000x128 f16)
        attn_wmma<<<BATCH * NHEADS * (TPAD / 16), 32, 0, stream>>>(
            bufA, rel + l * (2 * MAXLEN - 1), bufB);

        // o = attn @ wo[l] + bo[l]           (16000x128 @ 128x128) -> bufA f16
        gemm_f16_wmma<<<mp * (DMODEL / 16), 256, 0, stream>>>(
            bufB, wo16 + (size_t)l * DMODEL * DMODEL, bo + l * DMODEL,
            bufA, NTOK, DMODEL, DMODEL, 0);

        add_ln_kernel<<<NTOK, 128, 0, stream>>>(h, h16, bufA,
            ln1_g + l * DMODEL, ln1_b + l * DMODEL);

        // ffmid = relu(h @ w1[l] + b1[l])    (16000x128 @ 128x512) -> bufA f16
        gemm_f16_wmma<<<mp * (FFDIM / 16), 256, 0, stream>>>(
            h16, w116 + (size_t)l * DMODEL * FFDIM, b1 + l * FFDIM,
            bufA, NTOK, FFDIM, DMODEL, 1);

        // ffout = ffmid @ w2[l] + b2[l]      (16000x512 @ 512x128) -> bufB f16
        gemm_f16_wmma<<<mp * (DMODEL / 16), 256, 0, stream>>>(
            bufA, w216 + (size_t)l * FFDIM * DMODEL, b2 + l * DMODEL,
            bufB, NTOK, DMODEL, FFDIM, 0);

        add_ln_kernel<<<NTOK, 128, 0, stream>>>(h, h16, bufB,
            ln2_g + l * DMODEL, ln2_b + l * DMODEL);
    }

    pool_cls_kernel<<<BATCH, 128, 0, stream>>>(
        h, final_g, final_b, cls_w, cls_b, (float*)d_out);
}
